// Attention_77386720739372
// MI455X (gfx1250) — matrix operand order repaired and checked by reading, not compiled, not run on hardware
//
#include <hip/hip_runtime.h>

// ---------------------------------------------------------------------------
// Types for CDNA5 WMMA (wave32): V_WMMA_F32_16X16X32_BF16
// ---------------------------------------------------------------------------
typedef __bf16 bf16;
typedef __attribute__((ext_vector_type(16))) __bf16 v16bf;
typedef __attribute__((ext_vector_type(8)))  __bf16 v8bf;
typedef __attribute__((ext_vector_type(8)))  float  v8f;

union V16 { v16bf v; v8bf h[2]; };

static __device__ __forceinline__ bf16 f2bf(float f) {
  unsigned u = __builtin_bit_cast(unsigned, f);
  u += 0x7FFFu + ((u >> 16) & 1u);              // round-to-nearest-even
  return __builtin_bit_cast(bf16, (unsigned short)(u >> 16));
}
static __device__ __forceinline__ float bf2f(bf16 b) {
  unsigned u = ((unsigned)__builtin_bit_cast(unsigned short, b)) << 16;
  return __builtin_bit_cast(float, u);
}

// Async DMA: copy 32 contiguous bytes global -> LDS per lane (2 x B128).
// INST_OFFSET is added to BOTH the LDS and the global address, so offset:32
// covers the second half with the same address VGPRs.  Tracked by ASYNCcnt.
#define ASYNC_COPY32(ldsOff, gptr)                                            \
  asm volatile("global_load_async_to_lds_b128 %0, %1, off\n\t"                \
               "global_load_async_to_lds_b128 %0, %1, off offset:32"          \
               :: "v"(ldsOff), "v"(gptr) : "memory")

#define WAIT_ASYNC0() asm volatile("s_wait_asynccnt 0" ::: "memory")

static __device__ __forceinline__ unsigned lds_off(const void* p) {
  return (unsigned)(unsigned long long)p;   // low 32 bits = LDS byte offset
}

// ---------------------------------------------------------------------------
// Kernel 1: fp32 -> bf16 conversion
// ---------------------------------------------------------------------------
__global__ __launch_bounds__(256) void f32_to_bf16_kernel(
    const float* __restrict__ src, bf16* __restrict__ dst, int n) {
  int i = blockIdx.x * 256 + threadIdx.x;
  if (i < n) dst[i] = f2bf(src[i]);
}

// ---------------------------------------------------------------------------
// Kernel 2: TN GEMM, C[M,N] = A[M,K] * B[N,K]^T   (A,B bf16 row-major, K contig)
// 256 thr = 8 waves, tile 128(M) x 128(N), K-step 32, double-buffered LDS
// filled by global_load_async_to_lds_b128 (one barrier per k-step).
// Wave w: rows 32*(w>>1), cols 64*(w&1); 2x4 grid of 16x16 WMMA accumulators.
// ---------------------------------------------------------------------------
#define LDT 40   // LDS row stride (elems) for 32-wide K tiles; 80B, 16B aligned

__global__ __launch_bounds__(256) void gemm_bf16_tn(
    const bf16* __restrict__ A, const bf16* __restrict__ B,
    bf16* __restrict__ Cb, float* __restrict__ Cf, int M, int N, int K) {
  __shared__ bf16 As[2][128 * LDT];
  __shared__ bf16 Bs[2][128 * LDT];

  const int tid  = threadIdx.x;
  const int lane = tid & 31;
  const int wave = tid >> 5;
  const int wm   = wave >> 1;          // 0..3
  const int wn   = wave & 1;           // 0..1
  const int m0   = blockIdx.y * 128;
  const int n0   = blockIdx.x * 128;

  const int l15 = lane & 15;
  const int hi  = lane >> 4;
  const int kbA = hi ? 8 : 0;          // A frag K base (16-bit A 16x32 layout)
  const int kbB = hi ? 8 : 0;          // B frag K base (16-bit B 32x16 layout)

  const int sRow = tid >> 1;           // staging: 128 rows x 32 K, 16 elem/thr
  const int sCol = (tid & 1) * 16;

  const bf16* ag = A + (size_t)(m0 + sRow) * K + sCol;
  const bf16* bg = B + (size_t)(n0 + sRow) * K + sCol;
  unsigned la[2], lb[2];
  la[0] = lds_off(&As[0][sRow * LDT + sCol]);
  la[1] = lds_off(&As[1][sRow * LDT + sCol]);
  lb[0] = lds_off(&Bs[0][sRow * LDT + sCol]);
  lb[1] = lds_off(&Bs[1][sRow * LDT + sCol]);

  // prologue: stage k-step 0 into buffer 0
  ASYNC_COPY32(la[0], ag);
  ASYNC_COPY32(lb[0], bg);

  v8f acc[2][4];
#pragma unroll
  for (int i = 0; i < 2; i++)
#pragma unroll
    for (int j = 0; j < 4; j++)
#pragma unroll
      for (int r = 0; r < 8; r++) acc[i][j][r] = 0.0f;

  const int steps = K >> 5;
  for (int s = 0; s < steps; s++) {
    const int cur = s & 1;
    WAIT_ASYNC0();                 // this wave's DMA into buf[cur] landed
    __syncthreads();               // everyone's landed; buf[cur^1] free
    if (s + 1 < steps) {           // prefetch next k-step into other buffer
      ASYNC_COPY32(la[cur ^ 1], ag + (s + 1) * 32);
      ASYNC_COPY32(lb[cur ^ 1], bg + (s + 1) * 32);
    }

    V16 af[2], bfr[4];
#pragma unroll
    for (int mf = 0; mf < 2; mf++) {
      const bf16* p = &As[cur][(wm * 32 + mf * 16 + l15) * LDT];
      af[mf].h[0] = *(const v8bf*)(p + kbA);        // K = kb..kb+7
      af[mf].h[1] = *(const v8bf*)(p + 16 + kbA);   // K = 16+kb..16+kb+7
    }
#pragma unroll
    for (int nf = 0; nf < 4; nf++) {
      const bf16* p = &Bs[cur][(wn * 64 + nf * 16 + l15) * LDT] + kbB;
      bfr[nf].h[0] = *(const v8bf*)(p);
      bfr[nf].h[1] = *(const v8bf*)(p + 16);
    }
#pragma unroll
    for (int mf = 0; mf < 2; mf++)
#pragma unroll
      for (int nf = 0; nf < 4; nf++)
        acc[mf][nf] = __builtin_amdgcn_wmma_f32_16x16x32_bf16(
            false, af[mf].v, false, bfr[nf].v, (short)0, acc[mf][nf],
            false, false);
  }

#pragma unroll
  for (int mf = 0; mf < 2; mf++)
#pragma unroll
    for (int nf = 0; nf < 4; nf++)
#pragma unroll
      for (int r = 0; r < 8; r++) {
        int gm = m0 + wm * 32 + mf * 16 + r + 8 * hi;
        int gn = n0 + wn * 64 + nf * 16 + l15;
        float v = acc[mf][nf][r];
        if (Cb) Cb[(size_t)gm * N + gn] = f2bf(v);
        else    Cf[(size_t)gm * N + gn] = v;
      }
}

// ---------------------------------------------------------------------------
// Kernel 3: split qkv (bn,6144) -> Qh/Kh/Vh (bh,n,128), rotary on first 32
// dims of q,k; q scaled by d^-1/2.
// ---------------------------------------------------------------------------
__global__ __launch_bounds__(256) void rotary_split_kernel(
    const bf16* __restrict__ qkv, bf16* __restrict__ Qh,
    bf16* __restrict__ Kh, bf16* __restrict__ Vh) {
  int idx  = blockIdx.x * 256 + threadIdx.x;       // < 32*2048*128
  int dd   = idx & 127;
  int nidx = (idx >> 7) & 2047;
  int bh   = idx >> 18;
  int b = bh >> 4, hh = bh & 15;
  size_t srow = ((size_t)(b * 2048 + nidx)) * 6144 + hh * 128;
  float qv = bf2f(qkv[srow + dd]);
  float kv = bf2f(qkv[srow + 2048 + dd]);
  float vv = bf2f(qkv[srow + 4096 + dd]);
  if (dd < 32) {
    int j = dd & 15;                                // pos = concat(freqs,freqs)
    float inv = __expf(-(float)j * (9.2103403719761836f / 16.0f)); // 10000^(-j/16)
    float ang = (float)nidx * inv;
    float c = cosf(ang), s = sinf(ang);
    int   pdd = (dd < 16) ? dd + 16 : dd - 16;      // rotate_half partner
    float sgn = (dd < 16) ? -1.0f : 1.0f;
    float qp = bf2f(qkv[srow + pdd]);
    float kp = bf2f(qkv[srow + 2048 + pdd]);
    qv = qv * c + sgn * qp * s;
    kv = kv * c + sgn * kp * s;
  }
  qv *= 0.08838834764831845f;                        // 128^-0.5
  size_t drow = ((size_t)bh * 2048 + nidx) * 128 + dd;
  Qh[drow] = f2bf(qv);
  Kh[drow] = f2bf(kv);
  Vh[drow] = f2bf(vv);
}

// ---------------------------------------------------------------------------
// Kernel 4: causal flash attention.  Grid (n/128, 32 heads), 256 thr.
// Each wave owns 16 query rows; streams key/value tiles of Bc=32.
// K tile staged by async DMA; V tile transposed manually (DMA can't transpose).
// O written directly in (b, n, h*d) layout for the output projection.
// ---------------------------------------------------------------------------
__global__ __launch_bounds__(256) void flash_attn_kernel(
    const bf16* __restrict__ Q, const bf16* __restrict__ Kh,
    const bf16* __restrict__ Vh, bf16* __restrict__ O) {
  __shared__ bf16 Ks[32 * 136];        // K tile, row-major (key, d)
  __shared__ bf16 Vt[128 * 40];        // V tile transposed (d, key)
  __shared__ bf16 Ps[8 * 16 * 40];     // per-wave P scratch (16 x 32)

  const int blk = blockIdx.x;          // 128-row slab
  const int bh  = blockIdx.y;
  const int tid = threadIdx.x, lane = tid & 31, wave = tid >> 5;
  const int l15 = lane & 15, hi = lane >> 4;
  const int kbA = hi ? 8 : 0, kbB = hi ? 8 : 0;
  const int mBase = blk * 128 + wave * 16;
  const size_t headOff = (size_t)bh * 2048 * 128;

  // preload Q fragments: 16 rows x 128 d = 4 A-frags (16x32)
  V16 qf[4];
  const bf16* qrow = Q + headOff + (size_t)(mBase + l15) * 128;
#pragma unroll
  for (int dk = 0; dk < 4; dk++) {
    qf[dk].h[0] = *(const v8bf*)(qrow + dk * 32 + kbA);
    qf[dk].h[1] = *(const v8bf*)(qrow + dk * 32 + 16 + kbA);
  }

  v8f o[8];
  float mrun[8], lrun[8];
#pragma unroll
  for (int t = 0; t < 8; t++) {
#pragma unroll
    for (int r = 0; r < 8; r++) o[t][r] = 0.0f;
    mrun[t] = -__builtin_inff();
    lrun[t] = 0.0f;
  }

  // staging assignment: 32 keys x 128 d, 16 elems/thread
  const int sKey = tid >> 3;
  const int sD0  = (tid & 7) * 16;
  const unsigned ldsK = lds_off(&Ks[sKey * 136 + sD0]);

  const int nTiles = blk * 4 + 4;      // causal: keys up to (blk+1)*128
  for (int t = 0; t < nTiles; t++) {
    const int key0 = t * 32;
    {
      const bf16* kg = Kh + headOff + (size_t)(key0 + sKey) * 128 + sD0;
      const bf16* vg = Vh + headOff + (size_t)(key0 + sKey) * 128 + sD0;
      v8bf v0 = *(const v8bf*)vg, v1 = *(const v8bf*)(vg + 8);
      __syncthreads();                 // prev iteration done reading tiles
      ASYNC_COPY32(ldsK, kg);          // DMA K tile row into LDS
#pragma unroll
      for (int j = 0; j < 8; j++) {    // scatter V transposed
        Vt[(sD0 + j) * 40 + sKey]     = v0[j];
        Vt[(sD0 + 8 + j) * 40 + sKey] = v1[j];
      }
      WAIT_ASYNC0();
      __syncthreads();
    }

    // S = Q * K^T : two 16x16 f32 tiles (keys 0..15, 16..31 of the tile)
    v8f s0, s1;
#pragma unroll
    for (int r = 0; r < 8; r++) { s0[r] = 0.0f; s1[r] = 0.0f; }
#pragma unroll
    for (int dk = 0; dk < 4; dk++) {
      V16 b0, b1;
      const bf16* p0 = &Ks[l15 * 136 + dk * 32 + kbB];
      const bf16* p1 = &Ks[(16 + l15) * 136 + dk * 32 + kbB];
      b0.h[0] = *(const v8bf*)p0; b0.h[1] = *(const v8bf*)(p0 + 16);
      b1.h[0] = *(const v8bf*)p1; b1.h[1] = *(const v8bf*)(p1 + 16);
      s0 = __builtin_amdgcn_wmma_f32_16x16x32_bf16(false, qf[dk].v, false, b0.v,
                                                   (short)0, s0, false, false);
      s1 = __builtin_amdgcn_wmma_f32_16x16x32_bf16(false, qf[dk].v, false, b1.v,
                                                   (short)0, s1, false, false);
    }

    // causal mask + online softmax (row = r + 8*hi within wave's 16 rows)
    bf16* pw = &Ps[wave * 16 * 40];
#pragma unroll
    for (int r = 0; r < 8; r++) {
      int m  = mBase + r + 8 * hi;
      if (key0 + l15 > m)      s0[r] = -3.0e38f;
      if (key0 + 16 + l15 > m) s1[r] = -3.0e38f;
      float v = fmaxf(s0[r], s1[r]);
      v = fmaxf(v, __shfl_xor(v, 1, 16));
      v = fmaxf(v, __shfl_xor(v, 2, 16));
      v = fmaxf(v, __shfl_xor(v, 4, 16));
      v = fmaxf(v, __shfl_xor(v, 8, 16));
      float mnew  = fmaxf(mrun[r], v);
      float alpha = __expf(mrun[r] - mnew);
      mrun[r] = mnew;
      float p0 = __expf(s0[r] - mnew);
      float p1 = __expf(s1[r] - mnew);
      float ps = p0 + p1;
      ps += __shfl_xor(ps, 1, 16);
      ps += __shfl_xor(ps, 2, 16);
      ps += __shfl_xor(ps, 4, 16);
      ps += __shfl_xor(ps, 8, 16);
      lrun[r] = lrun[r] * alpha + ps;
#pragma unroll
      for (int nf = 0; nf < 8; nf++) o[nf][r] *= alpha;
      int M = r + 8 * hi;               // stash P for layout change C->A
      pw[M * 40 + l15]      = f2bf(p0);
      pw[M * 40 + 16 + l15] = f2bf(p1);
    }
    asm volatile("s_wait_dscnt 0" ::: "memory");   // wave-local LDS ordering

    V16 pf;                             // reload P as 16x32 A fragment
    const bf16* pr = &pw[l15 * 40 + kbA];
    pf.h[0] = *(const v8bf*)pr;
    pf.h[1] = *(const v8bf*)(pr + 16);

    // O += P * V  (V transposed in LDS -> contiguous B fragments)
#pragma unroll
    for (int nf = 0; nf < 8; nf++) {
      V16 vf;
      const bf16* vp = &Vt[(nf * 16 + l15) * 40 + kbB];
      vf.h[0] = *(const v8bf*)vp; vf.h[1] = *(const v8bf*)(vp + 16);
      o[nf] = __builtin_amdgcn_wmma_f32_16x16x32_bf16(false, pf.v, false, vf.v,
                                                      (short)0, o[nf], false, false);
    }
  }

  // normalize and store in (b, n, h*d) layout
  const int b = bh >> 4, hh = bh & 15;
#pragma unroll
  for (int r = 0; r < 8; r++) {
    float inv = 1.0f / lrun[r];
    int m = mBase + r + 8 * hi;
    bf16* orow = O + ((size_t)(b * 2048 + m)) * 2048 + hh * 128;
#pragma unroll
    for (int nf = 0; nf < 8; nf++)
      orow[nf * 16 + l15] = f2bf(o[nf][r] * inv);
  }
}

// ---------------------------------------------------------------------------
// Launcher
// ---------------------------------------------------------------------------
extern "C" void kernel_launch(void* const* d_in, const int* in_sizes, int n_in,
                              void* d_out, int out_size, void* d_ws, size_t ws_size,
                              hipStream_t stream) {
  (void)in_sizes; (void)n_in; (void)out_size; (void)ws_size;
  const float* x     = (const float*)d_in[0];   // (2,2048,2048)
  const float* w_qkv = (const float*)d_in[1];   // (6144,2048)
  const float* w_out = (const float*)d_in[2];   // (2048,2048)
  // d_in[3] (g) is unused by the reference computation.

  char* ws = (char*)d_ws;                        // ~168 MB used
  bf16* xb   = (bf16*)(ws);                      // 16 MiB
  bf16* wqb  = (bf16*)(ws + 16777216);           // 24 MiB
  bf16* wob  = (bf16*)(ws + 41943040);           //  8 MiB
  bf16* qkvb = (bf16*)(ws + 50331648);           // 48 MiB
  bf16* Qh   = (bf16*)(ws + 100663296);          // 16 MiB
  bf16* Kh   = (bf16*)(ws + 117440512);          // 16 MiB
  bf16* Vh   = (bf16*)(ws + 134217728);          // 16 MiB
  bf16* Ob   = (bf16*)(ws + 150994944);          // 16 MiB

  f32_to_bf16_kernel<<<8388608 / 256, 256, 0, stream>>>(x, xb, 8388608);
  f32_to_bf16_kernel<<<12582912 / 256, 256, 0, stream>>>(w_qkv, wqb, 12582912);
  f32_to_bf16_kernel<<<4194304 / 256, 256, 0, stream>>>(w_out, wob, 4194304);

  // qkv = x @ w_qkv^T   (M=4096, N=6144, K=2048)
  gemm_bf16_tn<<<dim3(6144 / 128, 4096 / 128), 256, 0, stream>>>(
      xb, wqb, qkvb, nullptr, 4096, 6144, 2048);

  rotary_split_kernel<<<(32 * 2048 * 128) / 256, 256, 0, stream>>>(qkvb, Qh, Kh, Vh);

  flash_attn_kernel<<<dim3(2048 / 128, 32), 256, 0, stream>>>(Qh, Kh, Vh, Ob);

  // out = attn_out @ w_out^T  (M=4096, N=2048, K=2048), fp32 result
  gemm_bf16_tn<<<dim3(2048 / 128, 4096 / 128), 256, 0, stream>>>(
      Ob, wob, nullptr, (float*)d_out, 4096, 2048, 2048);
}
